// Decoder_63642825392669
// MI455X (gfx1250) — compile-verified, matrix operand order checked by reading, not run on hardware
//
#include <hip/hip_runtime.h>

typedef float v2f __attribute__((ext_vector_type(2)));
typedef float v8f __attribute__((ext_vector_type(8)));

#define ZB   2
#define NPT  48
#define NY   49
#define MAXI 8
#define MAXP 176
#define LN2F 0.69314718055994531f

// ---------------- static layer configuration (constexpr-built) ----------------
struct PathT { int io, ii, lf, lo, li, mo, mi, w0, nrm; };
struct LayerCfg {
  int n_in, n_conv, n_paths;
  int dim_in, dim_conv, dim_out, n_weights, n_scal, n_gate;
  int in_mul[MAXI], in_l[MAXI], in_off[MAXI];
  int cv_mul[MAXI], cv_l[MAXI], cv_off[MAXI];
  PathT paths[MAXP];
};

constexpr LayerCfg makeLayer(int L) {
  LayerCfg c{};
  int mid[7] = {16, 16, 12, 8, 4, 4, 5};
  if (L == 0) { c.n_in = 1; c.in_mul[0] = 8; c.in_l[0] = 0; }
  else { c.n_in = 7; for (int l = 0; l < 7; l++) { c.in_mul[l] = mid[l]; c.in_l[l] = l; } }
  if (L < 2) {
    c.n_conv = 7; c.n_scal = 16; c.n_gate = 49;
    int cm[7] = {65, 16, 12, 8, 4, 4, 5};
    for (int l = 0; l < 7; l++) { c.cv_mul[l] = cm[l]; c.cv_l[l] = l; }
  } else {
    c.n_conv = 5; c.n_scal = 4; c.n_gate = 16;
    int cm[5] = {20, 4, 4, 4, 4};
    for (int l = 0; l < 5; l++) { c.cv_mul[l] = cm[l]; c.cv_l[l] = l; }
  }
  int off = 0;
  for (int i = 0; i < c.n_in; i++) { c.in_off[i] = off; off += c.in_mul[i] * (2 * c.in_l[i] + 1); }
  c.dim_in = off;
  off = 0;
  for (int i = 0; i < c.n_conv; i++) { c.cv_off[i] = off; off += c.cv_mul[i] * (2 * c.cv_l[i] + 1); }
  c.dim_conv = off;
  c.dim_out = c.dim_conv - c.n_gate;
  int npt[MAXI] = {};
  int np = 0, w = 0;
  for (int io = 0; io < c.n_conv; io++)
    for (int ii = 0; ii < c.n_in; ii++) {
      int lo = c.cv_l[io], li = c.in_l[ii];
      int lfmax = lo + li; if (lfmax > 6) lfmax = 6;
      int lfmin = lo - li; if (lfmin < 0) lfmin = -lfmin;
      for (int lf = lfmin; lf <= lfmax; lf++) {
        c.paths[np].io = io; c.paths[np].ii = ii; c.paths[np].lf = lf;
        c.paths[np].lo = lo; c.paths[np].li = li;
        c.paths[np].mo = c.cv_mul[io]; c.paths[np].mi = c.in_mul[ii];
        c.paths[np].w0 = w;
        w += c.cv_mul[io] * c.in_mul[ii];
        npt[io]++; np++;
      }
    }
  c.n_paths = np; c.n_weights = w;
  for (int p = 0; p < np; p++) c.paths[p].nrm = c.paths[p].mi * npt[c.paths[p].io];
  return c;
}

__constant__ LayerCfg c_cfg[3] = { makeLayer(0), makeLayer(1), makeLayer(2) };
static constexpr LayerCfg h_cfg[3] = { makeLayer(0), makeLayer(1), makeLayer(2) };

struct W3Off { int off[7][7][7]; int total; };
constexpr W3Off makeW3Off() {
  W3Off w{}; int o = 0;
  for (int a = 0; a < 7; a++)
    for (int b = 0; b < 7; b++)
      for (int cc = 0; cc < 7; cc++) {
        int lo = a - b; if (lo < 0) lo = -lo;
        if (cc >= lo && cc <= a + b) { w.off[a][b][cc] = o; o += (2*a+1)*(2*b+1)*(2*cc+1); }
        else w.off[a][b][cc] = -1;
      }
  w.total = o; return w;
}
__constant__ W3Off c_w3o = makeW3Off();
static constexpr W3Off h_w3o = makeW3Off();

// ---------------- WMMA helper: D = A(16x4,f32) * B(4x16,f32) + C ----------------
#if __has_builtin(__builtin_amdgcn_wmma_f32_16x16x4_f32)
__device__ __forceinline__ v8f wmma4(v8f c, v2f a, v2f b) {
  return __builtin_amdgcn_wmma_f32_16x16x4_f32(false, a, false, b, (short)0, c, false, false);
}
#else
typedef _Float16 v16h __attribute__((ext_vector_type(16)));
__device__ __forceinline__ v8f wmma4(v8f c, v2f a, v2f b) {
  v16h ah = {0,0,0,0,0,0,0,0,0,0,0,0,0,0,0,0};
  v16h bh = {0,0,0,0,0,0,0,0,0,0,0,0,0,0,0,0};
  ah[0] = (_Float16)a[0]; ah[1] = (_Float16)a[1];
  bh[0] = (_Float16)b[0]; bh[1] = (_Float16)b[1];
  return __builtin_amdgcn_wmma_f32_16x16x32_f16(false, ah, false, bh, (short)0, c, false, false);
}
#endif

// branchless softplus via fast transcendental units (v_exp_f32 / v_log_f32 + cndmask)
__device__ __forceinline__ float splus(float x) {
  float xc = fminf(x, 15.0f);
  float y = __logf(1.0f + __expf(xc));
  return (x > 15.0f) ? x : y;
}
__device__ __forceinline__ v8f vzero8() {
  v8f z = {0.f, 0.f, 0.f, 0.f, 0.f, 0.f, 0.f, 0.f};
  return z;
}

// ---------------- real Wigner-3j setup (device, double precision) ----------------
__device__ double dev_w3j(int j1, int j2, int j3, int m1, int m2, int m3, const double* F) {
  if (m1 + m2 + m3 != 0) return 0.0;
  int d = j1 - j2; if (d < 0) d = -d;
  if (j3 < d || j3 > j1 + j2) return 0.0;
  int a1 = m1 < 0 ? -m1 : m1, a2 = m2 < 0 ? -m2 : m2, a3 = m3 < 0 ? -m3 : m3;
  if (a1 > j1 || a2 > j2 || a3 > j3) return 0.0;
  int t0 = 0;
  if (j2 - j3 - m1 > t0) t0 = j2 - j3 - m1;
  if (j1 - j3 + m2 > t0) t0 = j1 - j3 + m2;
  int t1 = j1 + j2 - j3;
  if (j1 - m1 < t1) t1 = j1 - m1;
  if (j2 + m2 < t1) t1 = j2 + m2;
  double s = 0.0;
  for (int t = t0; t <= t1; t++) {
    double den = F[t] * F[j3 - j2 + m1 + t] * F[j3 - j1 - m2 + t] *
                 F[j1 + j2 - j3 - t] * F[j1 - m1 - t] * F[j2 + m2 - t];
    s += ((t & 1) ? -1.0 : 1.0) / den;
  }
  if ((j1 - j2 - m3) % 2) s = -s;
  s *= sqrt(F[j1 + j2 - j3] * F[j1 - j2 + j3] * F[j2 + j3 - j1] / F[j1 + j2 + j3 + 1]);
  s *= sqrt(F[j1 + a1] * F[j1 - a1] * F[j2 + a2] * F[j2 - a2] * F[j3 + a3] * F[j3 - a3]);
  return s;
}

__device__ int urow(int l, int mr, int* cm, double* cr, double* ci) {
  const double is2 = 0.70710678118654752440;
  if (mr < 0) {
    cm[0] = mr;  cr[0] = 0.0; ci[0] = is2;
    cm[1] = -mr; cr[1] = 0.0; ci[1] = (((-mr) & 1) ? is2 : -is2);
    return 2;
  } else if (mr == 0) {
    cm[0] = 0; cr[0] = 1.0; ci[0] = 0.0; return 1;
  }
  cm[0] = -mr; cr[0] = is2; ci[0] = 0.0;
  cm[1] = mr;  cr[1] = ((mr & 1) ? -is2 : is2); ci[1] = 0.0;
  return 2;
}

__global__ void setup_w3j_kernel(float* __restrict__ w3) {
  double F[20]; F[0] = 1.0;
  for (int i = 1; i < 20; i++) F[i] = F[i - 1] * (double)i;
  for (int l1 = 0; l1 < 7; l1++)
    for (int l2 = 0; l2 < 7; l2++) {
      int lmin = l1 - l2; if (lmin < 0) lmin = -lmin;
      int lmax = l1 + l2; if (lmax > 6) lmax = 6;
      for (int l3 = lmin; l3 <= lmax; l3++) {
        int n1 = 2 * l1 + 1, n2 = 2 * l2 + 1, n3 = 2 * l3 + 1;
        int off = c_w3o.off[l1][l2][l3];
        int useIm = (l1 + l2 + l3) & 1;
        int n = n1 * n2 * n3;
        for (int e = threadIdx.x; e < n; e += blockDim.x) {
          int i = e / (n2 * n3);
          int j = (e / n3) % n2;
          int k = e % n3;
          int cm1[2], cm2[2], cm3[2];
          double r1[2], i1[2], r2[2], i2[2], r3[2], i3[2];
          int q1 = urow(l1, i - l1, cm1, r1, i1);
          int q2 = urow(l2, j - l2, cm2, r2, i2);
          int q3 = urow(l3, k - l3, cm3, r3, i3);
          double re = 0.0, im = 0.0;
          for (int a = 0; a < q1; a++)
            for (int b = 0; b < q2; b++)
              for (int cc = 0; cc < q3; cc++) {
                double w = dev_w3j(l1, l2, l3, cm1[a], cm2[b], cm3[cc], F);
                if (w == 0.0) continue;
                double xr = r1[a] * r2[b] - i1[a] * i2[b];
                double xi = r1[a] * i2[b] + i1[a] * r2[b];
                double yr = xr * r3[cc] - xi * i3[cc];
                double yi = xr * i3[cc] + xi * r3[cc];
                re += yr * w; im += yi * w;
              }
          w3[off + e] = (float)(useIm ? im : re);
        }
      }
    }
}

// ---------------- fused layer kernel: one workgroup per (z, a) ----------------
__global__ void __launch_bounds__(256) conv_kernel(
    int layer, const float* __restrict__ geo, const float* __restrict__ fin,
    const float* __restrict__ W0, const float* __restrict__ b0,
    const float* __restrict__ W1, const float* __restrict__ b1,
    const float* __restrict__ W2, const float* __restrict__ b2,
    const float* __restrict__ w3, float* __restrict__ fout)
{
  extern __shared__ float sm[];
  const LayerCfg& C = c_cfg[layer];
  const int zz = blockIdx.x / NPT;
  const int apt = blockIdx.x % NPT;
  const int tid = threadIdx.x;
  const int lid = tid & 31;
  const int wv = tid >> 5;           // 8 wave32s
  const int lrow = lid & 15;
  const int lhi = lid >> 4;
  const int kbase = 2 * lhi;         // per-lane K sub-offset inside a WMMA step
  const int dim_in = C.dim_in;
  const int nw = C.n_weights;
  const float invsq = rsqrtf((float)NPT);   // folded into Y (messages are bilinear in Y*f)

  float* sY  = sm;                   // [48][49]  spherical harmonics * N^-1/2
  float* sRB = sY + NPT * NY;        // [48][12]  radial cosine basis (K-padded)
  float* sH1 = sRB + NPT * 12;       // [48][64]  radial hidden 1
  float* sH2 = sH1 + NPT * 64;       // [48][64]  radial hidden 2
  float* sF  = sH2 + NPT * 64;       // [48][dim_in] neighbor features (raw copy)
  float* sMp = sF + NPT * dim_in;    // [48][<=64] CG-coupled messages (chunked)
  float* sMs = sMp + NPT * 64;       // [64]      sum_b M (for bias term)
  float* sG  = sMs + 64;             // [64][<=64] G = H^T M
  float* sO  = sG + 64 * 64;         // [dim_conv] conv accumulator

  // ---- stage 0: async-stage the neighbor feature block (global -> LDS, ASYNCcnt) ----
#if __has_builtin(__builtin_amdgcn_global_load_async_to_lds_b128)
  {
    typedef int v4i __attribute__((vector_size(16)));
    typedef __attribute__((address_space(1))) v4i g_v4i;   // global
    typedef __attribute__((address_space(3))) v4i l_v4i;   // LDS
    const float* src = fin + zz * NPT * dim_in;   // contiguous [48*dim_in] block
    int nquads = (NPT * dim_in) >> 2;
    for (int idx = tid; idx < nquads; idx += 256) {
      __builtin_amdgcn_global_load_async_to_lds_b128(
          (g_v4i*)(src + 4 * idx),
          (l_v4i*)(sF + 4 * idx),
          0, 0);
    }
#if __has_builtin(__builtin_amdgcn_s_wait_asynccnt)
    __builtin_amdgcn_s_wait_asynccnt(0);
#else
    asm volatile("s_wait_asynccnt 0" ::: "memory");
#endif
  }
#else
  for (int idx = tid; idx < NPT * dim_in; idx += 256)
    sF[idx] = fin[zz * NPT * dim_in + idx];
#endif

  // ---- stage 1: geometry -> Y, radial basis (one thread per neighbor b) ----
  if (tid < NPT) {
    int bb = tid;
    float ax = geo[(zz * NPT + apt) * 3 + 0];
    float ay = geo[(zz * NPT + apt) * 3 + 1];
    float az = geo[(zz * NPT + apt) * 3 + 2];
    float rx = geo[(zz * NPT + bb) * 3 + 0] - ax;
    float ry = geo[(zz * NPT + bb) * 3 + 1] - ay;
    float rz = geo[(zz * NPT + bb) * 3 + 2] - az;
    float r = sqrtf(rx * rx + ry * ry + rz * rz + 1e-12f);
    float inv = 1.f / r;
    float ux = rx * inv, uy = ry * inv, uz = rz * inv;
    float msk = (r > 1e-6f) ? 1.f : 0.f;
    float Am[7], Bm[7];
    Am[0] = 1.f; Bm[0] = 0.f;
    for (int m = 1; m <= 6; m++) {
      float na = Am[m - 1] * ux - Bm[m - 1] * uy;
      float nb = Am[m - 1] * uy + Bm[m - 1] * ux;
      Am[m] = na; Bm[m] = nb;
    }
    float P[7][7];
    P[0][0] = 1.f;
    for (int m = 1; m <= 6; m++) P[m][m] = -(float)(2 * m - 1) * P[m - 1][m - 1];
    for (int m = 0; m < 6; m++) P[m + 1][m] = (float)(2 * m + 1) * uz * P[m][m];
    for (int m = 0; m <= 6; m++)
      for (int l = m + 2; l <= 6; l++)
        P[l][m] = ((float)(2 * l - 1) * uz * P[l - 1][m] - (float)(l + m - 1) * P[l - 2][m]) / (float)(l - m);
    double fct[14]; fct[0] = 1.0;
    for (int t = 1; t <= 13; t++) fct[t] = fct[t - 1] * (double)t;
    const float SQ2 = 1.41421356237309515f;
    int yo = 0;
    for (int l = 0; l <= 6; l++)
      for (int m = -l; m <= l; m++) {
        int am = m < 0 ? -m : m;
        float nrm = (float)sqrt((2.0 * l + 1.0) / (4.0 * 3.14159265358979323846) * fct[l - am] / fct[l + am]);
        float val;
        if (m == 0)      val = nrm * P[l][0];
        else if (m > 0)  val = SQ2 * nrm * P[l][am] * Am[am];
        else             val = SQ2 * nrm * P[l][am] * Bm[am];
        if (l > 0) val *= msk;
        sY[bb * NY + yo] = val * invsq;   // carries the per-layer 1/sqrt(N)
        yo++;
      }
    const float step = 5.0f / 9.0f;
    for (int k = 0; k < 10; k++) {
      float dd = (r - step * (float)k) / step;
      float h = 0.f;
      if (fabsf(dd) < 1.f) { float cv = cosf(1.57079632679489662f * dd); h = cv * cv; }
      sRB[bb * 12 + k] = h;
    }
    sRB[bb * 12 + 10] = 0.f; sRB[bb * 12 + 11] = 0.f;
  }
  for (int idx = tid; idx < C.dim_conv; idx += 256) sO[idx] = 0.f;
  __syncthreads();

  // ---- stage 2: radial MLP via WMMA. H1 = sp(RB@W0+b0)-ln2 ; H2 = sp(H1@W1+b1)-ln2 ----
  for (int t = wv; t < 12; t += 8) {          // 3 M-tiles x 4 N-tiles
    int mt = t >> 2, nt = t & 3;
    v8f acc = vzero8();
    for (int k0 = 0; k0 < 12; k0 += 4) {
      v2f av, bv;
#pragma unroll
      for (int q = 0; q < 2; q++) {
        int kk = k0 + kbase + q;
        av[q] = sRB[(mt * 16 + lrow) * 12 + kk];
        float wv0 = W0[(kk < 10 ? kk : 0) * 64 + nt * 16 + lrow];
        bv[q] = (kk < 10) ? wv0 : 0.f;
      }
      acc = wmma4(acc, av, bv);
    }
#pragma unroll
    for (int r = 0; r < 8; r++) {
      int row = mt * 16 + r + 8 * lhi;
      int col = nt * 16 + lrow;
      sH1[row * 64 + col] = splus(acc[r] + b0[col]) - LN2F;
    }
  }
  __syncthreads();
  for (int t = wv; t < 12; t += 8) {
    int mt = t >> 2, nt = t & 3;
    v8f acc = vzero8();
    for (int k0 = 0; k0 < 64; k0 += 4) {
      v2f av, bv;
#pragma unroll
      for (int q = 0; q < 2; q++) {
        int kk = k0 + kbase + q;
        av[q] = sH1[(mt * 16 + lrow) * 64 + kk];
        bv[q] = W1[kk * 64 + nt * 16 + lrow];
      }
      acc = wmma4(acc, av, bv);
    }
#pragma unroll
    for (int r = 0; r < 8; r++) {
      int row = mt * 16 + r + 8 * lhi;
      int col = nt * 16 + lrow;
      sH2[row * 64 + col] = splus(acc[r] + b1[col]) - LN2F;
    }
  }
  __syncthreads();

  // ---- stage 3: per-path coupled contraction, rank-64 factorized ----
  for (int p = 0; p < C.n_paths; p++) {
    PathT pt = C.paths[p];
    const int lo = pt.lo, li = pt.li, lf = pt.lf;
    const int mo = pt.mo, mi = pt.mi;
    const int no = 2 * lo + 1, ni = 2 * li + 1, nf = 2 * lf + 1;
    const int foff = C.in_off[pt.ii];
    const int yoff = lf * lf;
    const int coff = C.cv_off[pt.io];
    const int w0 = pt.w0;
    const float* Cw = w3 + c_w3o.off[lo][li][lf];
    const float nrm = rsqrtf((float)pt.nrm);
    __builtin_prefetch(W2 + w0, 0, 1);
    int vch = 64 / no; if (vch > mi) vch = mi;
    const int nut = (mo + 15) >> 4;
    v8f c2 = vzero8();                        // wave wv owns output u-tile wv (if < nut)
    for (int v0 = 0; v0 < mi; v0 += vch) {
      int nv = mi - v0; if (nv > vch) nv = vch;
      int nvi = nv * no;
      int nvi16 = (nvi + 15) & ~15;
      // (a) messages M[b,(v,i)] = sum_{j,k} C[i,j,k] Y[b,k] f[b,v,j]
      for (int idx = tid; idx < NPT * nvi16; idx += 256) {
        int bb = idx / nvi16, cc = idx - bb * nvi16;
        float m = 0.f;
        if (cc < nvi) {
          int v = cc / no;
          int i = cc - v * no;
          const float* fb = &sF[bb * dim_in + foff + (v0 + v) * ni];
          const float* yb = &sY[bb * NY + yoff];
          for (int j = 0; j < ni; j++) {
            const float* crow = &Cw[(i * ni + j) * nf];
            float s = 0.f;
            for (int k = 0; k < nf; k++) s += crow[k] * yb[k];
            m += s * fb[j];
          }
        }
        sMp[idx] = m;
      }
      __syncthreads();
      // (a2) Msum for bias term + GEMM1: G[h,(v,i)] = sum_b H2[b,h] M[b,(v,i)]
      for (int cc = tid; cc < nvi16; cc += 256) {
        float s = 0.f;
        for (int bb = 0; bb < NPT; bb++) s += sMp[bb * nvi16 + cc];
        sMs[cc] = s;
      }
      int ntn = nvi16 >> 4;
      for (int t = wv; t < 4 * ntn; t += 8) {
        int mt = t / ntn, nt = t - mt * ntn;
        v8f acc = vzero8();
        for (int k0 = 0; k0 < NPT; k0 += 4) {
          v2f av, bv;
#pragma unroll
          for (int q = 0; q < 2; q++) {
            int kk = k0 + kbase + q;                   // kk = neighbor b
            av[q] = sH2[kk * 64 + mt * 16 + lrow];     // A[h, b] = H2[b, h]
            bv[q] = sMp[kk * nvi16 + nt * 16 + lrow];  // B[b, (v,i)]
          }
          acc = wmma4(acc, av, bv);
        }
#pragma unroll
        for (int r = 0; r < 8; r++)
          sG[(mt * 16 + r + 8 * lhi) * 64 + nt * 16 + lrow] = acc[r];
      }
      __syncthreads();
      // (c) GEMM2 partial: o[u,i] += sum_{h,v in chunk} W2[h,(u,v)] G[h,(v,i)]
      // K ordering is vl-major: kappa = vl*64 + h  ->  h = kappa & 63, vl = kappa >> 6
      if (wv < nut) {
        int u = wv * 16 + lrow;
        int uc = (u < mo) ? u : 0;                     // clamped (safe) address
        const float* W2u = W2 + w0 + uc * mi + v0;
        int gcol = (lrow < no) ? lrow : 0;             // clamped LDS column
        for (int k0 = 0; k0 < 64 * nv; k0 += 4) {
          v2f av, bv;
#pragma unroll
          for (int q = 0; q < 2; q++) {
            int kk = k0 + kbase + q;
            int h = kk & 63, vl = kk >> 6;
            float wval = W2u[h * nw + vl];
            av[q] = (u < mo) ? wval : 0.f;
            float gval = sG[h * 64 + vl * no + gcol];
            bv[q] = (lrow < no) ? gval : 0.f;
          }
          c2 = wmma4(c2, av, bv);
        }
      }
      // bias part of R: sum_v b2[(u,v)] * sum_b M[b,(v,i)]
      for (int idx = tid; idx < mo * no; idx += 256) {
        int u = idx / no, i = idx - (idx / no) * no;
        float s = 0.f;
        for (int vl = 0; vl < nv; vl++)
          s += b2[w0 + u * mi + v0 + vl] * sMs[vl * no + i];
        sO[coff + idx] += nrm * s;
      }
      __syncthreads();
    }
    if (wv < nut) {
#pragma unroll
      for (int r = 0; r < 8; r++) {
        int u = wv * 16 + r + 8 * lhi;
        if (u < mo && lrow < no) sO[coff + u * no + lrow] += nrm * c2[r];
      }
    }
    __syncthreads();
  }

  // ---- stage 4: gated nonlinearity, write features ----
  float* op = fout + (zz * NPT + apt) * C.dim_out;
  const int ns = C.n_scal;
  for (int i = tid; i < ns; i += 256) op[i] = fmaxf(sO[i], 0.f);
  const int totnon = C.dim_out - ns;
  for (int idx = tid; idx < totnon; idx += 256) {
    int rem = idx, gb = 0;
    for (int io = 1; io < C.n_conv; io++) {
      int m = C.cv_mul[io], nn = 2 * C.cv_l[io] + 1, d = m * nn;
      if (rem < d) {
        int u = rem / nn;
        float g = 1.f / (1.f + __expf(-sO[ns + gb + u]));
        op[ns + idx] = sO[C.cv_off[io] + rem] * g;
        break;
      }
      rem -= d; gb += m;
    }
  }
}

// ---------------- launch ----------------
extern "C" void kernel_launch(void* const* d_in, const int* in_sizes, int n_in,
                              void* d_out, int out_size, void* d_ws, size_t ws_size,
                              hipStream_t stream) {
  const float* feat = (const float*)d_in[0];
  const float* geo  = (const float*)d_in[1];
  float* ws = (float*)d_ws;
  float* w3  = ws;
  float* fb0 = ws + h_w3o.total;
  float* fb1 = fb0 + ZB * NPT * 325;

  (void)hipFuncSetAttribute(reinterpret_cast<const void*>(conv_kernel),
                            hipFuncAttributeMaxDynamicSharedMemorySize, 160 * 1024);

  setup_w3j_kernel<<<dim3(1), dim3(256), 0, stream>>>(w3);

  for (int L = 0; L < 3; L++) {
    const LayerCfg& C = h_cfg[L];
    size_t smem = (size_t)(NPT * NY + NPT * 12 + NPT * 64 * 3 + NPT * C.dim_in +
                           64 + 64 * 64 + C.dim_conv) * sizeof(float);
    const float* fin = (L == 0) ? feat : (L == 1 ? fb0 : fb1);
    float* fout = (L == 0) ? fb0 : (L == 1 ? fb1 : (float*)d_out);
    int base = 2 + L * 6;   // pytree: features, geometry, then per-layer Ws[0..2], bs[0..2]
    conv_kernel<<<dim3(ZB * NPT), dim3(256), smem, stream>>>(
        L, geo, fin,
        (const float*)d_in[base + 0], (const float*)d_in[base + 3],
        (const float*)d_in[base + 1], (const float*)d_in[base + 4],
        (const float*)d_in[base + 2], (const float*)d_in[base + 5],
        w3, fout);
  }
  (void)in_sizes; (void)n_in; (void)out_size; (void)ws_size;
}
